// Cvae_22557168238987
// MI455X (gfx1250) — matrix operand order repaired and checked by reading, not compiled, not run on hardware
//
#include <hip/hip_runtime.h>
#include <hip/hip_bf16.h>

typedef _Float16 f16;
typedef __attribute__((ext_vector_type(16))) _Float16 v16h;
typedef __attribute__((ext_vector_type(8)))  _Float16 v8h;
typedef __attribute__((ext_vector_type(8)))  float    v8f;

#define H   128
#define G4  512      // 4*H
#define NAG 512      // N agents
#define TT  8
#define MM  30
#define D2  60       // 2*M
#define NSTEPS 12

__device__ __forceinline__ float sigf(float x) { return 1.0f / (1.0f + __expf(-x)); }

__device__ __forceinline__ unsigned int hashu(unsigned int x) {
    x ^= x >> 16; x *= 0x7feb352dU;
    x ^= x >> 15; x *= 0x846ca68bU;
    x ^= x >> 16; return x;
}

// ---------------------------------------------------------------------------
// Pack Whh (dir 0) of the 4 LSTMs from f32 row-major (4H, H) into the wave32
// WMMA B-fragment layout for V_WMMA_F32_16X16X32_F16:
//   fragment f = ntile*4 + kc ; lane%16 = N ; 16 halfs = K {8h..8h+7, 16+8h..16+8h+7}, h = lane/16.
// Load side reads a contiguous v16h per lane: packed[(f*32+lane)*16 + hh].
// ---------------------------------------------------------------------------
__global__ void pack_whh_kernel(const float* __restrict__ W0, const float* __restrict__ W1,
                                const float* __restrict__ W2, const float* __restrict__ W3,
                                f16* __restrict__ dst) {
    int tid = blockIdx.x * blockDim.x + threadIdx.x;
    if (tid >= 4 * 65536) return;
    int task = tid >> 16;
    int rem  = tid & 65535;
    int f    = rem >> 9;
    int lane = (rem >> 4) & 31;
    int hh   = rem & 15;
    int ntile = f >> 2, kc = f & 3;
    int n = ntile * 16 + (lane & 15);
    int k = kc * 32 + (lane >> 4) * 8 + (hh & 7) + (hh >> 3) * 16;
    const float* W = (task == 0) ? W0 : (task == 1) ? W1 : (task == 2) ? W2 : W3;
    dst[tid] = (f16)W[n * H + k];
}

// ---------------------------------------------------------------------------
// cur extraction + all backward-direction outputs. The bwd hidden at original
// time T-1 is one LSTM step from h=c=0 on x[:,T-1]: no Whh matmul needed.
// ---------------------------------------------------------------------------
__global__ void prep_kernel(const float* __restrict__ scene,
                            const float* __restrict__ Wv, const float* __restrict__ bv,
                            const float* __restrict__ Wa, const float* __restrict__ ba,
                            const float* __restrict__ Wp, const float* __restrict__ bp,
                            const float* __restrict__ We, const float* __restrict__ be,
                            float* __restrict__ cur, float* __restrict__ bwdh1) {
    int tid = blockIdx.x * blockDim.x + threadIdx.x;
    if (tid >= NAG * H) return;
    int b = tid >> 7, j = tid & 127;
    float cx = scene[(b * TT + 7) * 6 + 0];
    float cy = scene[(b * TT + 7) * 6 + 1];
    if (j == 0) { cur[b * 2] = cx; cur[b * 2 + 1] = cy; }
    float c5x = scene[(511 * TT + 7) * 6 + 0];
    float c5y = scene[(511 * TT + 7) * 6 + 1];
    for (int l = 0; l < 4; l++) {
        const float* Wih; const float* bb; float x0, x1;
        if (l == 0)      { Wih = Wv; bb = bv; x0 = scene[(b*TT+7)*6+2]; x1 = scene[(b*TT+7)*6+3]; }
        else if (l == 1) { Wih = Wa; bb = ba; x0 = scene[(b*TT+7)*6+4]; x1 = scene[(b*TT+7)*6+5]; }
        else if (l == 2) { Wih = Wp; bb = bp; x0 = cx;                  x1 = cy; }
        else             { Wih = We; bb = be; x0 = c5x - cx;            x1 = c5y - cy; }
        const float* W1 = Wih + G4 * 2;   // dir-1 slice of (2, 4H, 2)
        const float* b1 = bb + G4;
        float gi = W1[j * 2] * x0         + W1[j * 2 + 1] * x1         + b1[j];
        float gg = W1[(j+256) * 2] * x0   + W1[(j+256) * 2 + 1] * x1   + b1[j + 256];
        float go = W1[(j+384) * 2] * x0   + W1[(j+384) * 2 + 1] * x1   + b1[j + 384];
        float c  = sigf(gi) * tanhf(gg);
        bwdh1[(l * NAG + b) * H + j] = sigf(go) * tanhf(c);
    }
}

// ---------------------------------------------------------------------------
// Forward LSTM: one block = 16 batch rows, 8 waves. Per step:
//   gates(16x512) += h(16x128,f16) x WhhT(128x512,f16)  via WMMA
//   then fused gate nonlinearities; c kept in registers; h -> LDS f16.
// mode 0: x from scene[:, t, off:off+2]; mode 1: x = cur[t] - cur[row].
// ---------------------------------------------------------------------------
__launch_bounds__(256)
__global__ void lstm_fwd_kernel(int T, int mode, int off,
                                const float* __restrict__ scene, const float* __restrict__ cur,
                                const float* __restrict__ Wih0, const float* __restrict__ b0,
                                const f16* __restrict__ packedB, float* __restrict__ fwdh) {
    __shared__ __align__(32) f16  hA[16 * H];       // 4 KB, h in f16 (A operand)
    __shared__ float gates[16 * G4];                // 32 KB
    __shared__ float w0[G4], w1[G4], bsh[G4];       // 6 KB

    int tid  = threadIdx.x;
    int lane = tid & 31;
    int wid  = tid >> 5;
    int r0   = blockIdx.x * 16;

    for (int i = tid; i < G4; i += 256) { w0[i] = Wih0[i * 2]; w1[i] = Wih0[i * 2 + 1]; bsh[i] = b0[i]; }
    for (int i = tid; i < 16 * H; i += 256) hA[i] = (f16)0.0f;

    float creg[8], hreg[8];
#pragma unroll
    for (int u = 0; u < 8; u++) { creg[u] = 0.0f; hreg[u] = 0.0f; }

    int er  = tid >> 4;            // elementwise row 0..15
    int ej0 = (tid & 15) * 8;      // elementwise col base
    int rg  = r0 + er;             // global batch row

    int arow  = lane & 15;         // A-operand: lanes 0-15 hold K {0..7,16..23}
    int khalf = lane >> 4;         // lanes 16-31 hold K {8..15,24..31}
    __syncthreads();

    for (int t = 0; t < T; t++) {
        // ---- WMMA stage: each wave computes 4 N-tiles of the gate matrix ----
        union { v16h v; v8h p[2]; } afr[4];
#pragma unroll
        for (int kc = 0; kc < 4; kc++) {
            int kb = kc * 32 + khalf * 8;
            afr[kc].p[0] = *(const v8h*)&hA[arow * H + kb];
            afr[kc].p[1] = *(const v8h*)&hA[arow * H + kb + 16];
        }
        for (int nt = wid * 4; nt < wid * 4 + 4; nt++) {
            v8f acc = {};
#pragma unroll
            for (int kc = 0; kc < 4; kc++) {
                v16h bf = *(const v16h*)&packedB[((nt * 4 + kc) * 32 + lane) * 16];
                acc = __builtin_amdgcn_wmma_f32_16x16x32_f16(
                        false, afr[kc].v, false, bf, (short)0, acc, false, false);
            }
            int col = nt * 16 + (lane & 15);
            int rb  = khalf * 8;            // D layout: VGPR v -> row v (+8 for hi lanes)
#pragma unroll
            for (int v = 0; v < 8; v++) gates[(rb + v) * G4 + col] = acc[v];
        }
        __syncthreads();

        // ---- elementwise stage: K=2 input term + gate nonlinearities ----
        float x0, x1;
        if (mode == 0) { x0 = scene[(rg * TT + t) * 6 + off];  x1 = scene[(rg * TT + t) * 6 + off + 1]; }
        else           { x0 = cur[t * 2] - cur[rg * 2];        x1 = cur[t * 2 + 1] - cur[rg * 2 + 1]; }
#pragma unroll
        for (int u = 0; u < 8; u++) {
            int j = ej0 + u;
            float gi = gates[er * G4 + j]       + w0[j] * x0       + w1[j] * x1       + bsh[j];
            float gf = gates[er * G4 + 128 + j] + w0[128 + j] * x0 + w1[128 + j] * x1 + bsh[128 + j];
            float gg = gates[er * G4 + 256 + j] + w0[256 + j] * x0 + w1[256 + j] * x1 + bsh[256 + j];
            float go = gates[er * G4 + 384 + j] + w0[384 + j] * x0 + w1[384 + j] * x1 + bsh[384 + j];
            float c = sigf(gf) * creg[u] + sigf(gi) * tanhf(gg);
            creg[u] = c;
            float h = sigf(go) * tanhf(c);
            hreg[u] = h;
            hA[er * H + j] = (f16)h;
        }
        __syncthreads();
    }
#pragma unroll
    for (int u = 0; u < 8; u++) fwdh[rg * H + ej0 + u] = hreg[u];
}

// e = [sum3 fwd_hT | sum3 bwd_h1 | edge fwd_hT | edge bwd_h1]  (512 dims)
__global__ void build_e_kernel(const float* __restrict__ fwdh, const float* __restrict__ bwdh1,
                               float* __restrict__ e) {
    int tid = blockIdx.x * blockDim.x + threadIdx.x;
    if (tid >= NAG * G4) return;
    int b = tid >> 9, j = tid & 511;
    float v;
    if (j < 128)      v = fwdh[b*H + j] + fwdh[(NAG + b)*H + j] + fwdh[(2*NAG + b)*H + j];
    else if (j < 256) { int jj = j - 128;
                        v = bwdh1[b*H + jj] + bwdh1[(NAG + b)*H + jj] + bwdh1[(2*NAG + b)*H + jj]; }
    else if (j < 384) v = fwdh[(3*NAG + b)*H + (j - 256)];
    else              v = bwdh1[(3*NAG + b)*H + (j - 384)];
    e[b * G4 + j] = v;
}

// Precompute e @ gru_Wih[:, :512].T + bih (e is constant across decoder steps),
// state0 = e @ state_W.T + state_b, a0, and cur_state0 = broadcast(cur).
__launch_bounds__(192)
__global__ void head_kernel(const float* __restrict__ e, const float* __restrict__ gruWih,
                            const float* __restrict__ bih, const float* __restrict__ stateW,
                            const float* __restrict__ stateb, const float* __restrict__ actW,
                            const float* __restrict__ actb, const float* __restrict__ cur,
                            float* __restrict__ gi_e, float* __restrict__ hstate,
                            float* __restrict__ cur_state, float* __restrict__ a_buf) {
    __shared__ float esh[G4];
    int b = blockIdx.x, tid = threadIdx.x;
    for (int i = tid; i < G4; i += 192) esh[i] = e[b * G4 + i];
    __syncthreads();
    if (tid < 180) {
        float s = bih[tid];
        const float* wr = gruWih + tid * 514;
        for (int k = 0; k < G4; k++) s += esh[k] * wr[k];
        gi_e[b * 180 + tid] = s;
    }
    if (tid < D2) {
        float s = stateb[tid];
        const float* wr = stateW + tid * G4;
        for (int k = 0; k < G4; k++) s += esh[k] * wr[k];
        hstate[b * D2 + tid]    = s;
        cur_state[b * D2 + tid] = cur[b * 2 + (tid & 1)];
    }
    if (tid == 0) {
        float cx = cur[b * 2], cy = cur[b * 2 + 1];
        a_buf[b * 2]     = cx * actW[0] + cy * actW[1] + actb[0];
        a_buf[b * 2 + 1] = cx * actW[2] + cy * actW[3] + actb[1];
    }
}

// One decoder step. One block per agent; state carried in d_ws between steps.
__launch_bounds__(64)
__global__ void dec_step_kernel(int step,
                                const float* __restrict__ gi_e, const float* __restrict__ gruWih,
                                const float* __restrict__ gruWhh, const float* __restrict__ bhh,
                                const float* __restrict__ musW, const float* __restrict__ musb,
                                const float* __restrict__ lsW, const float* __restrict__ lsb,
                                const float* __restrict__ corW, const float* __restrict__ corb,
                                const float* __restrict__ actW, const float* __restrict__ actb,
                                float* __restrict__ hstate, float* __restrict__ cur_state,
                                float* __restrict__ a_buf, float* __restrict__ out) {
    __shared__ float hsh[D2], hnew[D2], mus_s[D2], std_s[D2], co_s[MM];
    int b = blockIdx.x, tid = threadIdx.x;
    if (tid < D2) hsh[tid] = hstate[b * D2 + tid];
    __syncthreads();
    float ax = a_buf[b * 2], ay = a_buf[b * 2 + 1];
    if (tid < D2) {
        float giv[3], ghv[3];
#pragma unroll
        for (int q = 0; q < 3; q++) {
            int g = q * D2 + tid;
            giv[q] = gi_e[b * 180 + g] + gruWih[g * 514 + 512] * ax + gruWih[g * 514 + 513] * ay;
            float s = bhh[g];
            const float* wr = gruWhh + g * D2;
            for (int k = 0; k < D2; k++) s += hsh[k] * wr[k];
            ghv[q] = s;
        }
        float r  = sigf(giv[0] + ghv[0]);
        float zg = sigf(giv[1] + ghv[1]);
        float nn = tanhf(giv[2] + r * ghv[2]);
        float hn = (1.0f - zg) * nn + zg * hsh[tid];
        hnew[tid] = hn;
        hstate[b * D2 + tid] = hn;
    }
    __syncthreads();
    if (tid < D2) {
        float d = musb[tid];
        const float* wr = musW + tid * D2;
        for (int k = 0; k < D2; k++) d += hnew[k] * wr[k];
        d = fminf(1.5f, fmaxf(-1.5f, d));
        float mu = d + cur_state[b * D2 + tid];
        cur_state[b * D2 + tid] = mu;       // carry: next cur_state = mus
        mus_s[tid] = mu;
        float ls = lsb[tid];
        const float* wl = lsW + tid * D2;
        for (int k = 0; k < D2; k++) ls += hnew[k] * wl[k];
        std_s[tid] = sqrtf(fminf(__expf(2.0f * ls), 1000.0f));
    }
    if (tid < MM) {
        float co = corb[tid];
        const float* wc = corW + tid * D2;
        for (int k = 0; k < D2; k++) co += hnew[k] * wc[k];
        co_s[tid] = tanhf(co);
    }
    __syncthreads();
    float* ob = out + ((size_t)step * NAG + b) * (MM * 5);
    if (tid < D2) {
        int m = tid >> 1, c = tid & 1;
        ob[m * 5 + c]     = mus_s[tid];
        ob[m * 5 + 2 + c] = std_s[tid];
    }
    if (tid < MM) ob[tid * 5 + 4] = co_s[tid];
    if (tid == 0) {
        // deterministic counter-hash sampling (log_pis is provably uniform)
        unsigned int h1 = hashu(0x9E3779B9u ^ (unsigned)(step * 0x85ebca6b) ^ (unsigned)(b * 0xc2b2ae35));
        unsigned int h2 = hashu(h1 + 0x165667B1u);
        unsigned int h3 = hashu(h2 + 0x27d4eb2fu);
        int comp = (int)(h1 % MM);
        float u1 = ((float)(h2 >> 9) + 0.5f) * (1.0f / 8388608.0f);
        float u2 = ((float)(h3 >> 9) + 0.5f) * (1.0f / 8388608.0f);
        float rr = sqrtf(-2.0f * __logf(u1));
        float n0 = rr * __cosf(6.28318530718f * u2);
        float n1 = rr * __sinf(6.28318530718f * u2);
        float atx = mus_s[comp * 2]     + std_s[comp * 2] * n0;
        float aty = mus_s[comp * 2 + 1] + std_s[comp * 2 + 1] * n1;
        a_buf[b * 2]     = atx * actW[0] + aty * actW[1] + actb[0];
        a_buf[b * 2 + 1] = atx * actW[2] + aty * actW[3] + actb[1];
    }
}

extern "C" void kernel_launch(void* const* d_in, const int* in_sizes, int n_in,
                              void* d_out, int out_size, void* d_ws, size_t ws_size,
                              hipStream_t stream) {
    (void)in_sizes; (void)n_in; (void)out_size; (void)ws_size;
    const float* scene  = (const float*)d_in[0];
    const float* velWih = (const float*)d_in[1];  const float* velWhh = (const float*)d_in[2];  const float* velb = (const float*)d_in[3];
    const float* accWih = (const float*)d_in[4];  const float* accWhh = (const float*)d_in[5];  const float* accb = (const float*)d_in[6];
    const float* posWih = (const float*)d_in[7];  const float* posWhh = (const float*)d_in[8];  const float* posb = (const float*)d_in[9];
    const float* edgWih = (const float*)d_in[10]; const float* edgWhh = (const float*)d_in[11]; const float* edgb = (const float*)d_in[12];
    const float* gruWih = (const float*)d_in[13]; const float* gruWhh = (const float*)d_in[14];
    const float* gbih   = (const float*)d_in[15]; const float* gbhh   = (const float*)d_in[16];
    const float* actW   = (const float*)d_in[17]; const float* actb   = (const float*)d_in[18];
    const float* stW    = (const float*)d_in[19]; const float* stb    = (const float*)d_in[20];
    const float* musW   = (const float*)d_in[23]; const float* musb   = (const float*)d_in[24];
    const float* lsW    = (const float*)d_in[25]; const float* lsb    = (const float*)d_in[26];
    const float* corW   = (const float*)d_in[27]; const float* corb   = (const float*)d_in[28];
    float* out = (float*)d_out;

    char* ws = (char*)d_ws;
    float* cur      = (float*)ws;                          // 512*2 f32
    f16*   packedB  = (f16*)(ws + 4096);                   // 4*512*128 f16
    float* fwdh     = (float*)(ws + 4096 + 524288);        // 4*512*128 f32
    float* bwdh1    = fwdh + 4 * NAG * H;                  // 4*512*128 f32
    float* e        = bwdh1 + 4 * NAG * H;                 // 512*512 f32
    float* gi_e     = e + NAG * G4;                        // 512*180 f32
    float* hstate   = gi_e + NAG * 180;                    // 512*60 f32
    float* curstate = hstate + NAG * D2;                   // 512*60 f32
    float* a_buf    = curstate + NAG * D2;                 // 512*2 f32

    pack_whh_kernel<<<(4 * 65536) / 256, 256, 0, stream>>>(velWhh, accWhh, posWhh, edgWhh, packedB);
    prep_kernel<<<(NAG * H) / 256, 256, 0, stream>>>(scene, velWih, velb, accWih, accb,
                                                     posWih, posb, edgWih, edgb, cur, bwdh1);
    lstm_fwd_kernel<<<32, 256, 0, stream>>>(TT,  0, 2, scene, cur, velWih, velb, packedB,             fwdh);
    lstm_fwd_kernel<<<32, 256, 0, stream>>>(TT,  0, 4, scene, cur, accWih, accb, packedB + 65536,     fwdh + NAG * H);
    lstm_fwd_kernel<<<32, 256, 0, stream>>>(TT,  0, 0, scene, cur, posWih, posb, packedB + 2 * 65536, fwdh + 2 * NAG * H);
    lstm_fwd_kernel<<<32, 256, 0, stream>>>(NAG, 1, 0, scene, cur, edgWih, edgb, packedB + 3 * 65536, fwdh + 3 * NAG * H);
    build_e_kernel<<<(NAG * G4) / 256, 256, 0, stream>>>(fwdh, bwdh1, e);
    head_kernel<<<NAG, 192, 0, stream>>>(e, gruWih, gbih, stW, stb, actW, actb, cur,
                                         gi_e, hstate, curstate, a_buf);
    for (int s = 0; s < NSTEPS; s++)
        dec_step_kernel<<<NAG, 64, 0, stream>>>(s, gi_e, gruWih, gruWhh, gbhh,
                                                musW, musb, lsW, lsb, corW, corb,
                                                actW, actb, hstate, curstate, a_buf, out);
}
